// GCNBaseNet_9569187136119
// MI455X (gfx1250) — matrix operand order repaired and checked by reading, not compiled, not run on hardware
//
#include <hip/hip_runtime.h>

#define G_  256
#define N_  60
#define NP_ 64
#define F_  128
#define D_  256
#define R_  4
#define E_  512
#define L_  3

typedef __attribute__((ext_vector_type(16))) __bf16 v16bf;
typedef __attribute__((ext_vector_type(8)))  float  v8f;

union FragAB { unsigned int u[8]; uint4 q[2]; v16bf v; };
union FragC  { float f[8]; v8f v; };

__device__ __forceinline__ unsigned short f2bf(float f) {
  unsigned int u = __float_as_uint(f);
  u += 0x7fffu + ((u >> 16) & 1u);           // round-to-nearest-even
  return (unsigned short)(u >> 16);
}
__device__ __forceinline__ float bf2f(unsigned short h) {
  return __uint_as_float(((unsigned int)h) << 16);
}

// CDNA5 async global->LDS copy (ASYNCcnt path), ISA §15.18.3 opcode 98.
// ldsoff = byte offset within wave's LDS (low 32 bits of flat shared addr).
__device__ __forceinline__ void async_ld_b128(unsigned ldsoff, const void* gaddr) {
  asm volatile("global_load_async_to_lds_b128 %0, %1, off"
               :: "v"(ldsoff), "v"(gaddr) : "memory");
}
__device__ __forceinline__ void wait_async(int n) {
  if (n == 0) asm volatile("s_wait_asynccnt 0x0" ::: "memory");
  else        asm volatile("s_wait_asynccnt 0x2" ::: "memory");
}

// ---------------- input padding / conversion ----------------
__global__ void pad_x_bf16(const float* __restrict__ x,
                           unsigned short* __restrict__ xp) {
  int i = blockIdx.x * blockDim.x + threadIdx.x;       // G*64*F
  int f = i % F_;
  int n = (i / F_) % NP_;
  int g = i / (F_ * NP_);
  float v = (n < N_) ? x[((size_t)g * N_ + n) * F_ + f] : 0.f;
  xp[i] = f2bf(v);
}

// Pre-swizzle f32 row-major W[K][N] into the WMMA-B operand layout:
// Bp[kt][n][p] (uint) = bf16(W[32kt+2p][n]) | bf16(W[32kt+2p+1][n])<<16
__global__ void pack_w_bf16(const float* __restrict__ W,
                            unsigned int* __restrict__ Bp, int K, int N) {
  int i = blockIdx.x * blockDim.x + threadIdx.x;       // K*N/2 uints
  if (i >= (K >> 1) * N) return;
  int p  = i & 15;
  int n  = (i >> 4) % N;
  int kt = i / (16 * N);
  int k  = (kt << 5) + (p << 1);
  unsigned int lo = f2bf(W[(size_t)k * N + n]);
  unsigned int hi = f2bf(W[(size_t)(k + 1) * N + n]);
  Bp[i] = lo | (hi << 16);
}

// ---------------- batched bf16 GEMM (WMMA, f32 accumulate) ----------------
// C[M,N] = A[M,K](bf16 row-major) x B (pre-swizzled Bp) (+bias)(+relu).
// Macro-tile 64x64, 8 waves, 2 WMMAs/wave per 32-K step. Double-buffered
// async global->LDS staging (ASYNCcnt) overlaps copy with matrix math.
// Requires M%64==0, N%64==0, K%32==0 (true for every call here).
__global__ __launch_bounds__(256)
void gemm_bf16_wmma(const unsigned short* __restrict__ A,
                    const unsigned int* __restrict__ Bp,
                    const float* __restrict__ bias,
                    unsigned short* __restrict__ Cb,   // bf16 out (or null)
                    float* __restrict__ Cf,            // f32 out  (or null)
                    int K, int lda, int N, int ldc, int tilesN,
                    unsigned long long sAz, unsigned long long sAy,
                    unsigned long long sBz, unsigned long long sBy,
                    unsigned long long sCz, unsigned long long sCy,
                    int relu)
{
  __shared__ unsigned int ldsA[2][64 * 16];  // 64 rows x 32 bf16, double-buffered
  __shared__ unsigned int ldsB[2][64 * 16];  // 64 cols x 16 packed K-pairs

  const int tid  = threadIdx.x;
  const int lane = tid & 31;
  const int wave = tid >> 5;
  const int wm   = (wave >> 1) << 4;       // 0,16,32,48
  const int wn2  = (wave & 1) << 5;        // 0,32  (two 16-col tiles each)
  const int half = lane >> 4;
  const int l15  = lane & 15;

  const int tileM = blockIdx.x / tilesN;
  const int tileN = blockIdx.x - tileM * tilesN;
  const int m0 = tileM * 64;
  const int n0 = tileN * 64;

  A  += (size_t)blockIdx.z * sAz + (size_t)blockIdx.y * sAy;
  Bp += (size_t)blockIdx.z * sBz + (size_t)blockIdx.y * sBy;
  const size_t cbase = (size_t)blockIdx.z * sCz + (size_t)blockIdx.y * sCy;

  FragC c0, c1;
#pragma unroll
  for (int i = 0; i < 8; ++i) { c0.f[i] = 0.f; c1.f[i] = 0.f; }

  // staging: thread `tid` owns one 16B chunk of each tile (64 rows x 4 uint4)
  const int srow = tid >> 2, sq = tid & 3;
  const unsigned aoff[2] = { (unsigned)(size_t)(&ldsA[0][0]) + (unsigned)tid * 16u,
                             (unsigned)(size_t)(&ldsA[1][0]) + (unsigned)tid * 16u };
  const unsigned boff[2] = { (unsigned)(size_t)(&ldsB[0][0]) + (unsigned)tid * 16u,
                             (unsigned)(size_t)(&ldsB[1][0]) + (unsigned)tid * 16u };
  const unsigned short* gA  = A  + (size_t)(m0 + srow) * lda + sq * 8;  // +32/tile
  const unsigned int*   gBt = Bp + ((size_t)(n0 + srow)) * 16 + sq * 4; // +N*16/tile

  const int KT = K >> 5;
  // prologue: start tile 0 into buffer 0
  async_ld_b128(aoff[0], gA);
  async_ld_b128(boff[0], gBt);

  for (int kt = 0; kt < KT; ++kt) {
    const int cur = kt & 1;
    if (kt + 1 < KT) {           // stream next tile into the other buffer
      async_ld_b128(aoff[cur ^ 1], gA + (size_t)(kt + 1) * 32);
      async_ld_b128(boff[cur ^ 1], gBt + (size_t)(kt + 1) * N * 16);
      wait_async(2);             // only this tile's 2 loads must have landed
    } else {
      wait_async(0);
    }
    __syncthreads();

    // A fragment (ISA 16-bit A 16x32): lane=M, VGPR0..3 K=half*8+{0..7},
    // VGPR4..7 K=16+half*8+{0..7} -> two b128 loads.
    FragAB a, b0, b1;
    const uint4* ap = reinterpret_cast<const uint4*>(&ldsA[cur][(wm + l15) * 16]);
    a.q[0] = ap[half];
    a.q[1] = ap[2 + half];
    // B fragments: lane=N, K-pairs pre-packed -> two b128 loads each.
    const uint4* bp0 = reinterpret_cast<const uint4*>(&ldsB[cur][(wn2 + l15) * 16]);
    b0.q[0] = bp0[half];
    b0.q[1] = bp0[2 + half];
    const uint4* bp1 = reinterpret_cast<const uint4*>(&ldsB[cur][(wn2 + 16 + l15) * 16]);
    b1.q[0] = bp1[half];
    b1.q[1] = bp1[2 + half];

    c0.v = __builtin_amdgcn_wmma_f32_16x16x32_bf16(false, a.v, false, b0.v,
                                                   (short)0, c0.v, false, false);
    c1.v = __builtin_amdgcn_wmma_f32_16x16x32_bf16(false, a.v, false, b1.v,
                                                   (short)0, c1.v, false, false);
    __syncthreads();             // buffer `cur` free for tile kt+2
  }

  // C/D layout: VGPR r -> M = r + half*8, lane&15 -> N
  const int gn0 = n0 + wn2 + l15;
  const int gn1 = gn0 + 16;
  const float bv0 = bias ? bias[gn0] : 0.f;
  const float bv1 = bias ? bias[gn1] : 0.f;
#pragma unroll
  for (int r = 0; r < 8; ++r) {
    const int gm = m0 + wm + (half << 3) + r;
    float v0 = c0.f[r] + bv0;
    float v1 = c1.f[r] + bv1;
    if (relu) { v0 = fmaxf(v0, 0.f); v1 = fmaxf(v1, 0.f); }
    const size_t ci = cbase + (size_t)gm * ldc;
    if (Cb) { Cb[ci + gn0] = f2bf(v0); Cb[ci + gn1] = f2bf(v1); }
    else    { Cf[ci + gn0] = v0;       Cf[ci + gn1] = v1; }
  }
}

// ---------------- GCN edge aggregation (one block per (g,r)) ----------------
__global__ __launch_bounds__(256)
void gcn_aggregate(const float* __restrict__ xw,
                   const int* __restrict__ edge_index,
                   const float* __restrict__ edge_weight,
                   const float* __restrict__ bias,
                   unsigned short* __restrict__ acc)
{
  extern __shared__ float smem[];
  float* xw_s  = smem;                         // 64*256 f32
  float* out_s = smem + NP_ * D_;              // 64*256 f32
  int*   er    = (int*)(out_s + NP_ * D_);     // E
  int*   ec    = er + E_;                      // E
  float* ew_s  = (float*)(ec + E_);            // E
  __shared__ float dis[NP_];

  const int r = blockIdx.x, g = blockIdx.y;
  const int tid = threadIdx.x;
  const size_t gr = (size_t)g * R_ + r;

  const float4* src = reinterpret_cast<const float4*>(xw + gr * NP_ * D_);
  for (int i = tid; i < NP_ * D_ / 4; i += 256) {
    reinterpret_cast<float4*>(xw_s)[i]  = src[i];
    reinterpret_cast<float4*>(out_s)[i] = make_float4(0.f, 0.f, 0.f, 0.f);
  }
  const int*   rowp = edge_index + gr * 2 * E_;
  const int*   colp = rowp + E_;
  const float* wp   = edge_weight + gr * E_;
  for (int e = tid; e < E_; e += 256) {
    er[e] = rowp[e]; ec[e] = colp[e]; ew_s[e] = wp[e];
  }
  if (tid < NP_) dis[tid] = (tid < N_) ? 1.f : 0.f;   // self-loop weight
  __syncthreads();

  for (int e = tid; e < E_; e += 256) atomicAdd(&dis[ec[e]], ew_s[e]);  // ds_add_f32
  __syncthreads();
  if (tid < NP_) {
    float d = dis[tid];
    dis[tid] = (d > 0.f) ? rsqrtf(d) : 0.f;
  }
  __syncthreads();

  // Each thread owns feature column `tid` -> race-free serial edge loop.
  for (int e = 0; e < E_; ++e) {
    const int rw = er[e], cl = ec[e];
    const float nrm = dis[rw] * ew_s[e] * dis[cl];
    out_s[cl * D_ + tid] += nrm * xw_s[rw * D_ + tid];
  }
  for (int n = 0; n < N_; ++n) {                      // self loops (w = 1)
    const float nrm = dis[n] * dis[n];
    out_s[n * D_ + tid] += nrm * xw_s[n * D_ + tid];
  }

  const float bb = bias[(size_t)r * D_ + tid];
  unsigned short* dst = acc + (size_t)g * NP_ * (R_ * D_) + (size_t)r * D_ + tid;
  for (int n = 0; n < NP_; ++n) {
    float v = (n < N_) ? (out_s[n * D_ + tid] + bb) : 0.f;
    dst[(size_t)n * (R_ * D_)] = f2bf(v);
  }
}

// ---------------- final tiny FC: [G,256] x [256,2] ----------------
__global__ __launch_bounds__(256)
void readout_fc3(const unsigned short* __restrict__ z2,
                 const float* __restrict__ W, const float* __restrict__ b,
                 float* __restrict__ out)
{
  __shared__ float r0[256], r1[256];
  const int g = blockIdx.x, t = threadIdx.x;
  const float p = bf2f(z2[(size_t)g * D_ + t]);
  r0[t] = p * W[t * 2 + 0];
  r1[t] = p * W[t * 2 + 1];
  __syncthreads();
  for (int s = 128; s > 0; s >>= 1) {
    if (t < s) { r0[t] += r0[t + s]; r1[t] += r1[t + s]; }
    __syncthreads();
  }
  if (t == 0) {
    out[g * 2 + 0] = r0[0] + b[0];
    out[g * 2 + 1] = r1[0] + b[1];
  }
}

// ---------------- host orchestration ----------------
extern "C" void kernel_launch(void* const* d_in, const int* in_sizes, int n_in,
                              void* d_out, int out_size, void* d_ws, size_t ws_size,
                              hipStream_t stream)
{
  (void)in_sizes; (void)n_in; (void)out_size; (void)ws_size;
  const float* x   = (const float*)d_in[0];
  const int*   ei  = (const int*)  d_in[1];
  const float* ew  = (const float*)d_in[2];
  const float* W0  = (const float*)d_in[3];
  const float* b0  = (const float*)d_in[4];
  const float* Wg  = (const float*)d_in[5];
  const float* bg  = (const float*)d_in[6];
  const float* Wi1 = (const float*)d_in[7];
  const float* bi1 = (const float*)d_in[8];
  const float* Wi2 = (const float*)d_in[9];
  const float* bi2 = (const float*)d_in[10];
  const float* Wf1 = (const float*)d_in[11];
  const float* bf1 = (const float*)d_in[12];
  const float* Wf2 = (const float*)d_in[13];
  const float* bf2 = (const float*)d_in[14];
  const float* Wf3 = (const float*)d_in[15];
  const float* bf3 = (const float*)d_in[16];
  float* out = (float*)d_out;

  char* ws = (char*)d_ws;
  size_t off = 0;
  auto alloc = [&](size_t bytes) -> char* {
    char* p = ws + off;
    off = (off + bytes + 255) & ~(size_t)255;
    return p;
  };
  typedef unsigned short bf;
  typedef unsigned int   u32;
  bf*    xpad = (bf*)alloc((size_t)G_ * NP_ * F_ * 2);
  u32*   W0p  = (u32*)alloc((size_t)R_ * F_ * D_ / 2 * 4);
  u32*   Wgp  = (u32*)alloc((size_t)(L_ - 1) * R_ * D_ * D_ / 2 * 4);
  u32*   Wi1p = (u32*)alloc((size_t)R_ * D_ * D_ / 2 * 4);
  u32*   Wi2p = (u32*)alloc((size_t)D_ * D_ / 2 * 4);
  u32*   Wf1p = (u32*)alloc((size_t)N_ * D_ * D_ / 2 * 4);
  u32*   Wf2p = (u32*)alloc((size_t)D_ * D_ / 2 * 4);
  float* xw   = (float*)alloc((size_t)G_ * R_ * NP_ * D_ * 4);
  bf*    accb = (bf*)alloc((size_t)G_ * NP_ * R_ * D_ * 2);
  bf*    tb   = (bf*)alloc((size_t)G_ * NP_ * D_ * 2);
  bf*    hb   = (bf*)alloc((size_t)G_ * NP_ * D_ * 2);   // [G][64][256] padded
  bf*    z1b  = (bf*)alloc((size_t)G_ * D_ * 2);
  bf*    z2b  = (bf*)alloc((size_t)G_ * D_ * 2);

  auto pack = [&](const float* s, u32* d, int K, int N) {
    int n = (K / 2) * N;
    pack_w_bf16<<<(n + 255) / 256, 256, 0, stream>>>(s, d, K, N);
  };
  pad_x_bf16<<<(G_ * NP_ * F_) / 256, 256, 0, stream>>>(x, xpad);
  pack(W0,  W0p,  R_ * F_, D_);
  pack(Wg,  Wgp,  (L_ - 1) * R_ * D_, D_);
  pack(Wi1, Wi1p, R_ * D_, D_);
  pack(Wi2, Wi2p, D_, D_);
  pack(Wf1, Wf1p, N_ * D_, D_);
  pack(Wf2, Wf2p, D_, D_);

  const size_t aggShm = (size_t)(2 * NP_ * D_) * 4 + (size_t)E_ * 12;

  for (int l = 0; l < L_; ++l) {
    const bf*  Aact = (l == 0) ? xpad : hb;
    const int  K1   = (l == 0) ? F_ : D_;
    const u32* Bw   = (l == 0) ? W0p : (Wgp + (size_t)(l - 1) * R_ * D_ * D_ / 2);
    // xw[g][r] = Aact[g] @ W[r]   (batched over z=g, y=r)
    gemm_bf16_wmma<<<dim3(D_ / 64, R_, G_), 256, 0, stream>>>(
        Aact, Bw, nullptr, nullptr, xw,
        K1, K1, D_, D_, D_ / 64,
        (unsigned long long)NP_ * K1, 0ull,
        0ull, (unsigned long long)(K1 * D_ / 2),
        (unsigned long long)R_ * NP_ * D_, (unsigned long long)NP_ * D_, 0);

    const float* brel = (l == 0) ? b0 : (bg + (size_t)(l - 1) * R_ * D_);
    gcn_aggregate<<<dim3(R_, G_), 256, aggShm, stream>>>(xw, ei, ew, brel, accb);

    // t = relu(acc @ Wi1 + bi1), M = G*64
    gemm_bf16_wmma<<<dim3((G_ * NP_ / 64) * (D_ / 64)), 256, 0, stream>>>(
        accb, Wi1p, bi1, tb, nullptr,
        R_ * D_, R_ * D_, D_, D_, D_ / 64,
        0ull, 0ull, 0ull, 0ull, 0ull, 0ull, 1);
    // h = t @ Wi2 + bi2
    gemm_bf16_wmma<<<dim3((G_ * NP_ / 64) * (D_ / 64)), 256, 0, stream>>>(
        tb, Wi2p, bi2, hb, nullptr,
        D_, D_, D_, D_, D_ / 64,
        0ull, 0ull, 0ull, 0ull, 0ull, 0ull, 0);
  }

  // z1 = relu(flat @ Wf1 + bf1): flat[g][k] = hb[g*16384 + k], k < 15360
  gemm_bf16_wmma<<<dim3((G_ / 64) * (D_ / 64)), 256, 0, stream>>>(
      hb, Wf1p, bf1, z1b, nullptr,
      N_ * D_, NP_ * D_, D_, D_, D_ / 64,
      0ull, 0ull, 0ull, 0ull, 0ull, 0ull, 1);
  // z2 = relu(z1 @ Wf2 + bf2)
  gemm_bf16_wmma<<<dim3((G_ / 64) * (D_ / 64)), 256, 0, stream>>>(
      z1b, Wf2p, bf2, z2b, nullptr,
      D_, D_, D_, D_, D_ / 64,
      0ull, 0ull, 0ull, 0ull, 0ull, 0ull, 1);
  // out = z2 @ Wf3 + bf3
  readout_fc3<<<G_, 256, 0, stream>>>(z2b, Wf3, bf3, out);
}
